// MambaWorldModel_28449863369031
// MI455X (gfx1250) — matrix operand order repaired and checked by reading, not compiled, not run on hardware
//
#include <hip/hip_runtime.h>
#include <hip/hip_bf16.h>

// ---------------------------------------------------------------------------
// Mamba world model forward, MI455X (gfx1250, wave32, WMMA + TDM).
// GEMMs: bf16 WMMA 16x16x32, f32 accumulate, 64x128 block tile.
//   - A tile: global_load_b128 f32 -> v_cvt bf16 -> ds_store_b64
//   - W tile: pre-transposed bf16 panels, loaded by tensor_load_to_lds (TDM)
//             with hardware LDS padding matching the 80B row stride.
// Scan: fused per-channel recurrence kernel.
// ---------------------------------------------------------------------------

typedef __attribute__((ext_vector_type(16))) __bf16          v16bf;
typedef __attribute__((ext_vector_type(8)))  float           v8f;
typedef __attribute__((ext_vector_type(4)))  float           f32x4;
typedef __attribute__((ext_vector_type(4)))  unsigned short  u16x4;
typedef __attribute__((ext_vector_type(8)))  unsigned short  u16x8;
typedef __attribute__((ext_vector_type(4)))  unsigned int    v4u;
typedef __attribute__((ext_vector_type(4)))  int             v4i;
typedef __attribute__((ext_vector_type(8)))  int             v8i;

typedef __attribute__((address_space(3))) unsigned short lds_us_t;

#define BM 64
#define BN 128
#define BK 32
#define LSTR 40   // LDS row stride in shorts: 80 B = 16B-aligned, bank-spread

__device__ __forceinline__ unsigned short f32_to_bf16_bits(float f) {
  return __builtin_bit_cast(unsigned short, (__bf16)f);   // v_cvt_*bf16_f32
}
__device__ __forceinline__ u16x4 cvt4(f32x4 v) {
  u16x4 r;
  r[0] = f32_to_bf16_bits(v[0]); r[1] = f32_to_bf16_bits(v[1]);
  r[2] = f32_to_bf16_bits(v[2]); r[3] = f32_to_bf16_bits(v[3]);
  return r;
}
__device__ __forceinline__ float activate(float v, int act) {
  if (act == 1) return v > 0.f ? v : 0.f;                 // relu
  if (act == 2) return v > 20.f ? v : log1pf(__expf(v));  // softplus
  return v;
}
__device__ __forceinline__ u16x8 lds_ld8(const unsigned short* p) {
  return *(const u16x8*)__builtin_assume_aligned((const void*)p, 16);
}
__device__ __forceinline__ u16x8 glb_ld8(const unsigned short* p) {
  return *(const u16x8*)__builtin_assume_aligned((const void*)p, 16);
}

// C[M,N] = act(A[M,K] @ W[K,N] + bias).
// A: f32 [M,K] stride lda. Wt: bf16, pre-transposed [Npad][K] (K contiguous).
// Requirements: M%64==0, K%32==0, N%16==0, Npad%128==0, lda%4==0.
__global__ __launch_bounds__(256) void gemm_bf16_wmma(
    const float* __restrict__ A, int lda,
    const unsigned short* __restrict__ Wt,
    const float* __restrict__ bias,
    float* __restrict__ C, int ldc,
    int M, int N, int Npad, int K, int act)
{
  __shared__ __align__(16) unsigned short sA [BM][LSTR];  // [row][k]
  __shared__ __align__(16) unsigned short sBt[BN][LSTR];  // [col][k] (transposed)

  const int tid  = threadIdx.x;
  const int lane = tid & 31;
  const int wave = tid >> 5;
  const int tm   = wave & 3;     // wave's 16-row tile
  const int tn   = wave >> 2;    // wave's 64-col half
  const int m    = lane & 15;    // matrix row (A) / col (B,C)
  const int h    = lane >> 4;    // lane-half select

  const int rowBase = blockIdx.x * BM;
  const int colBase = blockIdx.y * BN;

#if __has_builtin(__builtin_amdgcn_tensor_load_to_lds)
  const unsigned sBt_lds = (unsigned)(unsigned long long)(lds_us_t*)&sBt[0][0];
#endif

  v8f acc[4] = { {}, {}, {}, {} };

  for (int k0 = 0; k0 < K; k0 += BK) {
    // ---- W tile: 32(K) x 128(N) bf16 via Tensor Data Mover ----
#if __has_builtin(__builtin_amdgcn_tensor_load_to_lds)
    if (wave == 0) {
      unsigned long long ga =
          (unsigned long long)(const void*)(Wt + (long)colBase * K + k0);
      v4u g0;
      g0[0] = 1u;                                   // count=1 (valid descriptor)
      g0[1] = sBt_lds;                              // LDS dest address
      g0[2] = (unsigned)ga;                         // global_addr[31:0]
      g0[3] = (unsigned)((ga >> 32) & 0x01FFFFFFull) | (2u << 30);  // addr[56:32], type=2
      v8i g1;
      g1[0] = (1 << 16)      // data_size = 2 bytes
            | (1 << 20)      // pad_enable
            | (3 << 22)      // pad_interval: every 16 DWORDs (=64B tile row)
            | (3 << 25);     // pad_amount: 4 DWORDs (->80B LDS row = LSTR)
      g1[1] = (K & 0xFFFF) << 16;                           // tensor_dim0 lo
      g1[2] = ((unsigned)K >> 16) | ((Npad & 0xFFFF) << 16);// dim0 hi | dim1 lo
      g1[3] = ((unsigned)Npad >> 16) | (32 << 16);          // dim1 hi | tile_dim0=32
      g1[4] = 128;                                          // tile_dim1=128, tile_dim2=0
      g1[5] = K;                                            // tensor_dim0_stride lo32
      g1[6] = 0;                                            // stride hi | dim1_stride lo
      g1[7] = 0;
      v4i z4 = {0, 0, 0, 0};
  #if __has_include(<hip/amd_detail/amd_gfx1250_TDM.h>)
      v8i z8 = {0, 0, 0, 0, 0, 0, 0, 0};
      __builtin_amdgcn_tensor_load_to_lds(g0, g1, z4, z4, z8, 0);
  #else
      __builtin_amdgcn_tensor_load_to_lds(g0, g1, z4, z4, 0);
  #endif
  #if __has_builtin(__builtin_amdgcn_s_wait_tensorcnt)
      __builtin_amdgcn_s_wait_tensorcnt(0);
  #endif
    }
#else
    // fallback: wide synchronous copies (Wt rows are contiguous bf16)
    #pragma unroll
    for (int i = 0; i < 2; ++i) {
      int e = i * 256 + tid;
      int c = e >> 2, ch = e & 3;
      u16x8 v = glb_ld8(Wt + (long)(colBase + c) * K + k0 + 8 * ch);
      *(u16x8*)__builtin_assume_aligned((void*)&sBt[c][8 * ch], 16) = v;
    }
#endif

    // ---- A tile: 64x32 f32 as 512 float4 chunks (2 per thread) ----
    #pragma unroll
    for (int i = 0; i < 2; ++i) {
      int e = i * 256 + tid;
      int r = e >> 3, c4 = e & 7;
      f32x4 v = *(const f32x4*)(A + (long)(rowBase + r) * lda + k0 + 4 * c4);
      *(u16x4*)&sA[r][4 * c4] = cvt4(v);
    }
    // prefetch next A K-tile
    if (k0 + BK < K) {
      __builtin_prefetch(A + (long)(rowBase + (tid & 63)) * lda + k0 + BK, 0, 0);
    }
    __syncthreads();

    // ---- fragments: two ds_load_b128 each (ISA 7.12.2 16-bit layouts) ----
    union { u16x8 u[2]; v16bf v; } ua;
    ua.u[0] = lds_ld8(&sA[tm * 16 + m][h * 8]);        // K = h*8 + j
    ua.u[1] = lds_ld8(&sA[tm * 16 + m][16 + h * 8]);   // K = 16 + h*8 + j
    const v16bf af = ua.v;

    #pragma unroll
    for (int ct = 0; ct < 4; ++ct) {
      union { u16x8 u[2]; v16bf v; } ub;
      const unsigned short* bp = &sBt[tn * 64 + ct * 16 + m][h * 8];  // K = h*8 + j, 16 + h*8 + j
      ub.u[0] = lds_ld8(bp);
      ub.u[1] = lds_ld8(bp + 16);
      acc[ct] = __builtin_amdgcn_wmma_f32_16x16x32_bf16(
                    false, af, false, ub.v, (short)0, acc[ct], false, false);
    }
    __syncthreads();
  }

  // ---- epilogue: C/D layout row = 8*h + r, col = lane&15 ----
  const int gr0 = rowBase + tm * 16 + h * 8;
  #pragma unroll
  for (int ct = 0; ct < 4; ++ct) {
    int gc = colBase + tn * 64 + ct * 16 + m;
    if (gc < N) {
      float bv = bias ? bias[gc] : 0.f;
      #pragma unroll
      for (int r = 0; r < 8; ++r) {
        C[(long)(gr0 + r) * ldc + gc] = activate(acc[ct][r] + bv, act);
      }
    }
  }
}

// Wt[c][k] = bf16(W[k][c]) for c < N, 0 for N <= c < Npad
__global__ __launch_bounds__(256) void transpose_bf16_kernel(
    const float* __restrict__ W, unsigned short* __restrict__ Wt,
    int K, int N, int total /* = Npad*K */)
{
  int idx = blockIdx.x * 256 + threadIdx.x;
  if (idx >= total) return;
  int c = idx / K, k = idx - c * K;
  float v = (c < N) ? W[(long)k * N + c] : 0.f;
  Wt[idx] = f32_to_bf16_bits(v);
}

// cat[M,288] = [lat[M,256], action[M,32]]
__global__ __launch_bounds__(256) void concat_kernel(
    const float* __restrict__ lat, const float* __restrict__ actn,
    float* __restrict__ cat, int total)
{
  int idx = blockIdx.x * 256 + threadIdx.x;
  if (idx >= total) return;
  int row = idx / 288, c = idx - row * 288;
  cat[idx] = (c < 256) ? lat[(long)row * 256 + c]
                       : actn[(long)row * 32 + (c - 256)];
}

// depthwise causal conv1d (kernel 4) + bias + SiLU over x-half of xz
__global__ __launch_bounds__(256) void conv_silu_kernel(
    const float* __restrict__ xz, const float* __restrict__ cw,
    const float* __restrict__ cb, float* __restrict__ xc)
{
  int idx = blockIdx.x * 256 + threadIdx.x;   // idx = row*2048 + d
  int d   = idx & 2047;
  int row = idx >> 11;
  int t   = row & 1023;
  float acc = cb[d];
  #pragma unroll
  for (int k = 0; k < 4; ++k) {
    int tk = t + k - 3;
    if (tk >= 0) acc += xz[(long)(row + k - 3) * 4096 + d] * cw[d * 4 + k];
  }
  float s = 1.f / (1.f + __expf(-acc));
  xc[idx] = acc * s;
}

// selective scan: one thread per (b, d); h[16] in VGPRs; B_t/C_t via LDS.
// y_out = (sum_n h*C + Dp*x) * silu(z), fully fused.
__global__ __launch_bounds__(256) void scan_kernel(
    const float* __restrict__ xc,   const float* __restrict__ dtb,
    const float* __restrict__ dbc,  const float* __restrict__ xz,
    const float* __restrict__ A_log,const float* __restrict__ Dp,
    float* __restrict__ y)
{
  __shared__ float sBC[32];
  int idx = blockIdx.x * 256 + threadIdx.x;   // b*2048 + d ; block spans one b
  int d = idx & 2047;
  int b = idx >> 11;

  float A[16];
  #pragma unroll
  for (int n = 0; n < 16; ++n) A[n] = -__expf(A_log[d * 16 + n]);
  float Dv = Dp[d];
  float h[16];
  #pragma unroll
  for (int n = 0; n < 16; ++n) h[n] = 0.f;

  for (int t = 0; t < 1024; ++t) {
    int row = (b << 10) | t;
    if (threadIdx.x < 32) sBC[threadIdx.x] = dbc[(long)row * 96 + 64 + threadIdx.x];
    __syncthreads();
    float dtv = dtb[(long)row * 2048 + d];
    float xv  = xc [(long)row * 2048 + d];
    float zv  = xz [(long)row * 4096 + 2048 + d];
    float dx  = dtv * xv;
    float acc = 0.f;
    #pragma unroll
    for (int n = 0; n < 16; ++n) {
      h[n] = h[n] * __expf(dtv * A[n]) + dx * sBC[n];
      acc += h[n] * sBC[16 + n];
    }
    float yf = acc + Dv * xv;
    float sg = 1.f / (1.f + __expf(-zv));
    y[(long)row * 2048 + d] = yf * (zv * sg);
    __syncthreads();
  }
}

// reward head: wave per row, coalesced loads, wave32 shuffle reduction
__global__ __launch_bounds__(256) void rew_kernel(
    const float* __restrict__ h2, const float* __restrict__ w,
    const float* __restrict__ b,  float* __restrict__ out, int M)
{
  int lane = threadIdx.x & 31;
  int wv   = threadIdx.x >> 5;
  int row  = blockIdx.x * 8 + wv;
  if (row >= M) return;
  float acc = 0.f;
  #pragma unroll
  for (int i = 0; i < 8; ++i) {
    int j = lane + 32 * i;
    acc += h2[(long)row * 256 + j] * w[j];
  }
  #pragma unroll
  for (int off = 16; off > 0; off >>= 1) acc += __shfl_xor(acc, off);
  if (lane == 0) out[row] = acc + b[0];
}

extern "C" void kernel_launch(void* const* d_in, const int* in_sizes, int n_in,
                              void* d_out, int out_size, void* d_ws, size_t ws_size,
                              hipStream_t stream) {
  (void)in_sizes; (void)n_in; (void)out_size; (void)ws_size;
  const float* obs    = (const float*)d_in[0];   // (8,1024,128)
  const float* actn   = (const float*)d_in[1];   // (8,1024,32)
  const float* enc_W1 = (const float*)d_in[2];
  const float* enc_b1 = (const float*)d_in[3];
  const float* enc_W2 = (const float*)d_in[4];
  const float* enc_b2 = (const float*)d_in[5];
  const float* dyn_W  = (const float*)d_in[6];
  const float* dyn_b  = (const float*)d_in[7];
  const float* in_W   = (const float*)d_in[8];
  const float* conv_w = (const float*)d_in[9];
  const float* conv_b = (const float*)d_in[10];
  const float* x_W    = (const float*)d_in[11];
  const float* dt_W   = (const float*)d_in[12];
  const float* dt_b   = (const float*)d_in[13];
  const float* A_log  = (const float*)d_in[14];
  const float* Dp     = (const float*)d_in[15];
  const float* out_W  = (const float*)d_in[16];
  const float* obs_W1 = (const float*)d_in[17];
  const float* obs_b1 = (const float*)d_in[18];
  const float* obs_W2 = (const float*)d_in[19];
  const float* obs_b2 = (const float*)d_in[20];
  const float* rew_W1 = (const float*)d_in[21];
  const float* rew_b1 = (const float*)d_in[22];
  const float* rew_W2 = (const float*)d_in[23];
  const float* rew_b2 = (const float*)d_in[24];

  const int M = 8192;                 // B*L tokens
  float* ws  = (float*)d_ws;
  float* t0  = ws;                    // 8192x256  (enc hidden; later obs head hidden)
  float* lat = t0  + (long)M * 256;   // 8192x256  (later rew head hidden)
  float* cat = lat + (long)M * 256;   // 8192x288
  float* mi  = cat + (long)M * 288;   // 8192x1024
  float* xz  = mi  + (long)M * 1024;  // 8192x4096 (later reused for mo)
  float* xc  = xz  + (long)M * 4096;  // 8192x2048
  float* dbc = xc  + (long)M * 2048;  // 8192x96
  float* dtb = dbc + (long)M * 96;    // 8192x2048
  float* yb  = dtb + (long)M * 2048;  // 8192x2048
  float* mo  = xz;                    // xz dead after scan
  float* h1  = t0;
  float* h2  = lat;
  float* pred_obs = (float*)d_out;            // 8192x128
  float* pred_rew = pred_obs + (long)M * 128; // 8192x1

  // bf16 transposed weight panels (Npad x K each), after f32 workspace
  unsigned short* wtb      = (unsigned short*)(yb + (long)M * 2048);
  unsigned short* wtEncW1  = wtb;                            // 256 x 128
  unsigned short* wtEncW2  = wtEncW1 + 256 * 128;            // 256 x 256
  unsigned short* wtDynW   = wtEncW2 + 256 * 256;            // 1024 x 288
  unsigned short* wtInW    = wtDynW  + 1024 * 288;           // 4096 x 1024
  unsigned short* wtXW     = wtInW   + 4096 * 1024;          // 128 x 2048 (N=96 padded)
  unsigned short* wtDtW    = wtXW    + 128 * 2048;           // 2048 x 64
  unsigned short* wtOutW   = wtDtW   + 2048 * 64;            // 1024 x 2048
  unsigned short* wtObsW1  = wtOutW  + 1024 * 2048;          // 256 x 1024
  unsigned short* wtObsW2  = wtObsW1 + 256 * 1024;           // 128 x 256
  unsigned short* wtRewW1  = wtObsW2 + 128 * 256;            // 256 x 1024

  auto prep = [&](const float* W, unsigned short* Wt, int K, int N, int Npad) {
    int total = Npad * K;
    transpose_bf16_kernel<<<(total + 255) / 256, 256, 0, stream>>>(W, Wt, K, N, total);
  };
  auto gemm = [&](const float* A, int lda, const unsigned short* Wt,
                  const float* bias, float* C, int ldc, int N, int Npad,
                  int K, int act) {
    dim3 grid(M / BM, Npad / BN);
    hipLaunchKernelGGL(gemm_bf16_wmma, grid, dim3(256), 0, stream,
                       A, lda, Wt, bias, C, ldc, M, N, Npad, K, act);
  };

  // one-time (per launch) weight conversion/transposition to bf16 panels
  prep(enc_W1, wtEncW1, 128,  256,  256);
  prep(enc_W2, wtEncW2, 256,  256,  256);
  prep(dyn_W,  wtDynW,  288,  1024, 1024);
  prep(in_W,   wtInW,   1024, 4096, 4096);
  prep(x_W,    wtXW,    2048, 96,   128);
  prep(dt_W,   wtDtW,   64,   2048, 2048);
  prep(out_W,  wtOutW,  2048, 1024, 1024);
  prep(obs_W1, wtObsW1, 1024, 256,  256);
  prep(obs_W2, wtObsW2, 256,  128,  128);
  prep(rew_W1, wtRewW1, 1024, 256,  256);

  // encoder
  gemm(obs, 128, wtEncW1, enc_b1, t0, 256, 256, 256, 128, /*relu*/1);
  gemm(t0, 256, wtEncW2, enc_b2, lat, 256, 256, 256, 256, 0);
  // concat + dyn projection into d_model
  concat_kernel<<<(M * 288 + 255) / 256, 256, 0, stream>>>(lat, actn, cat, M * 288);
  gemm(cat, 288, wtDynW, dyn_b, mi, 1024, 1024, 1024, 288, 0);
  // mamba in_proj -> xz
  gemm(mi, 1024, wtInW, nullptr, xz, 4096, 4096, 4096, 1024, 0);
  // depthwise conv + silu
  conv_silu_kernel<<<(M * 2048) / 256, 256, 0, stream>>>(xz, conv_w, conv_b, xc);
  // x_proj -> (dt_in | B | C)
  gemm(xc, 2048, wtXW, nullptr, dbc, 96, 96, 128, 2048, 0);
  // dt = softplus(dt_in @ dt_W + dt_b)   (A = dbc first 64 cols, lda=96)
  gemm(dbc, 96, wtDtW, dt_b, dtb, 2048, 2048, 2048, 64, /*softplus*/2);
  // selective scan (fused D-skip + silu(z) gating)
  scan_kernel<<<(8 * 2048) / 256, 256, 0, stream>>>(xc, dtb, dbc, xz, A_log, Dp, yb);
  // out_proj
  gemm(yb, 2048, wtOutW, nullptr, mo, 1024, 1024, 1024, 2048, 0);
  // obs head
  gemm(mo, 1024, wtObsW1, obs_b1, h1, 256, 256, 256, 1024, 1);
  gemm(h1, 256, wtObsW2, obs_b2, pred_obs, 128, 128, 128, 256, 0);
  // reward head
  gemm(mo, 1024, wtRewW1, rew_b1, h2, 256, 256, 256, 1024, 1);
  rew_kernel<<<(M + 7) / 8, 256, 0, stream>>>(h2, rew_W2, rew_b2, pred_rew, M);
}
